// SelfAttention_807453852337
// MI455X (gfx1250) — matrix operand order repaired and checked by reading, not compiled, not run on hardware
//
#include <hip/hip_runtime.h>

// Self-attention (B=4, C=256, d=32, N=4096) — flash-attention style, f16 WMMA,
// fp32 accumulate, online softmax with wave-uniform max. gfx1250 wave32.

typedef __attribute__((ext_vector_type(16))) _Float16 v16h;
typedef __attribute__((ext_vector_type(8)))  _Float16 v8h;
typedef __attribute__((ext_vector_type(8)))  float    v8f;

#define NB 4
#define NC 256
#define ND 32     // qk feature dim (C/8)
#define NN 4096   // H*W
#define LOG2E 1.4426950408889634f

__device__ __forceinline__ v16h ld16h(const _Float16* p0, const _Float16* p1) {
  union { v16h v; v8h h[2]; } u;
  u.h[0] = *(const v8h*)p0;
  u.h[1] = *(const v8h*)p1;
  return u.v;
}

// ---------------------------------------------------------------------------
// Kernel 1: 1x1-conv projections  f = wf@x+bf, g = wg@x+bg, v = wh@x+bh
// Q is pre-scaled by log2(e) so softmax uses raw v_exp_f32 (exp2).
// Outputs f16:  Qh,Kh  as [B][N][32] (row-contig d)   Vh as [B][C][N]
// ---------------------------------------------------------------------------
__global__ __launch_bounds__(256) void proj_kernel(
    const float* __restrict__ x,
    const float* __restrict__ wf, const float* __restrict__ bf,
    const float* __restrict__ wg, const float* __restrict__ bg,
    const float* __restrict__ wh, const float* __restrict__ bh,
    _Float16* __restrict__ Qh, _Float16* __restrict__ Kh,
    _Float16* __restrict__ Vh)
{
  __shared__ float xs[NC][64];
  const int blk = blockIdx.x;            // 0..255
  const int b   = blk >> 6;
  const int n0  = (blk & 63) * 64;
  const int tid = threadIdx.x;
  const int p   = tid & 63;              // pixel within tile
  const int kg  = tid >> 6;              // 0..3

  const float* xb = x + ((size_t)b * NC) * NN + n0;
  #pragma unroll 4
  for (int cc = 0; cc < 64; ++cc) {
    int c = kg * 64 + cc;
    xs[c][p] = xb[(size_t)c * NN + p];
  }
  __syncthreads();

  // f and g: 8 output channels per thread
  for (int kk = 0; kk < 8; ++kk) {
    int k = kg * 8 + kk;
    const float* wfr = wf + k * NC;
    const float* wgr = wg + k * NC;
    float af = 0.f, ag = 0.f;
    #pragma unroll 4
    for (int c = 0; c < NC; ++c) {
      float xv = xs[c][p];
      af += wfr[c] * xv;
      ag += wgr[c] * xv;
    }
    af = (af + bf[k]) * LOG2E;           // fold log2(e) into Q
    ag += bg[k];
    size_t row = ((size_t)b * NN + n0 + p) * ND;
    Qh[row + k] = (_Float16)af;
    Kh[row + k] = (_Float16)ag;
  }
  // v: 64 output channels per thread
  for (int kk = 0; kk < 64; ++kk) {
    int k = kg * 64 + kk;
    const float* whr = wh + k * NC;
    float av = 0.f;
    #pragma unroll 4
    for (int c = 0; c < NC; ++c) av += whr[c] * xs[c][p];
    av += bh[k];
    Vh[((size_t)b * NC + k) * NN + n0 + p] = (_Float16)av;
  }
}

// ---------------------------------------------------------------------------
// Kernel 2: flash attention. One wave per (batch, 16-query tile, c-half).
// 64-key chunks: 4 score WMMAs (S^T = K x Q) + 16 PV WMMAs (out^T = V^T x P^T).
// Wave-uniform running max; accumulator rescale is branch-skipped when the
// max is unchanged (the common case), eliminating 64 v_mul per chunk.
// ---------------------------------------------------------------------------
__global__ __launch_bounds__(256) void attn_kernel(
    const _Float16* __restrict__ Qh, const _Float16* __restrict__ Kh,
    const _Float16* __restrict__ Vh, const float* __restrict__ x,
    const float* __restrict__ gamma_p, float* __restrict__ out)
{
  const int lane  = threadIdx.x & 31;
  const int wid   = threadIdx.x >> 5;
  const int tile  = blockIdx.x * 8 + wid;      // 0..2047
  const int chalf = tile & 1;                  // which 128 channels
  const int it    = tile >> 1;                 // 0..1023
  const int b     = it >> 8;                   // 0..3
  const int i0    = (it & 255) << 4;           // query tile base
  const bool hi   = lane >= 16;
  const int  l15  = lane & 15;
  const int  dseg = hi ? 8 : 0;

  // Q as B-fragment (32x16): lane holds column i=l15, K = dseg+e (e<8), 16+dseg+(e-8) (e>=8)
  const _Float16* qp = Qh + ((size_t)b * NN + i0 + l15) * ND + dseg;
  const v16h qB = ld16h(qp, qp + 16);

  v8f acc[8];
  const v8f vzero = {};
  #pragma unroll
  for (int t = 0; t < 8; ++t) acc[t] = vzero;

  float m = -3.0e38f;                          // wave-uniform running max
  float lsum = 0.f;

  const _Float16* kb = Kh + (size_t)b * NN * ND;
  const _Float16* vb = Vh + (size_t)b * NC * NN;

  for (int j0 = 0; j0 < NN; j0 += 64) {
    // --- K A-fragments (4 tiles of 16 keys), score WMMAs ---
    v8f s[4];
    #pragma unroll
    for (int jt = 0; jt < 4; ++jt) {
      const _Float16* kp = kb + (size_t)(j0 + jt * 16 + l15) * ND + dseg;
      v16h kA = ld16h(kp, kp + 16);
      s[jt] = __builtin_amdgcn_wmma_f32_16x16x32_f16(false, kA, false, qB,
                                                     (short)0, vzero, false, false);
    }

    // --- wave-uniform chunk max (butterfly over all 32 lanes) ---
    float cm = s[0][0];
    #pragma unroll
    for (int jt = 0; jt < 4; ++jt)
      #pragma unroll
      for (int e = 0; e < 8; ++e) cm = fmaxf(cm, s[jt][e]);
    #pragma unroll
    for (int w = 1; w < 32; w <<= 1) cm = fmaxf(cm, __shfl_xor(cm, w, 32));

    if (cm > m) {                              // uniform branch: rare after warm-up
      float scale = exp2f(m - cm);
      m = cm;
      lsum *= scale;
      #pragma unroll
      for (int t = 0; t < 8; ++t)
        #pragma unroll
        for (int e = 0; e < 8; ++e) acc[t][e] *= scale;
    }

    // --- exp2 (Q pre-scaled by log2e) and partial row sums ---
    float ls = 0.f;
    #pragma unroll
    for (int jt = 0; jt < 4; ++jt)
      #pragma unroll
      for (int e = 0; e < 8; ++e) {
        s[jt][e] = exp2f(s[jt][e] - m);
        ls += s[jt][e];
      }
    ls += __shfl_xor(ls, 16, 32);
    lsum += ls;

    // --- rebuild P^T as two B-fragments (32 keys x 16 queries) ---
    v16h pB0, pB1;
    #pragma unroll
    for (int e = 0; e < 8; ++e) {
      float a0 = __shfl_xor(s[0][e], 16, 32);
      float a1 = __shfl_xor(s[1][e], 16, 32);
      float b0 = __shfl_xor(s[2][e], 16, 32);
      float b1 = __shfl_xor(s[3][e], 16, 32);
      pB0[e]     = (_Float16)(hi ? a1 : s[0][e]);
      pB0[8 + e] = (_Float16)(hi ? s[1][e] : a0);
      pB1[e]     = (_Float16)(hi ? b1 : s[2][e]);
      pB1[8 + e] = (_Float16)(hi ? s[3][e] : b0);
    }

    // --- out^T += V^T x P^T : 8 channel tiles x 2 key sub-chunks ---
    #pragma unroll
    for (int t = 0; t < 8; ++t) {
      const _Float16* vp = vb + ((size_t)(chalf * 128 + t * 16 + l15)) * NN
                              + j0 + (hi ? 16 : 0);
      v16h vA0 = ld16h(vp, vp + 8);
      v16h vA1 = ld16h(vp + 32, vp + 40);
      acc[t] = __builtin_amdgcn_wmma_f32_16x16x32_f16(false, vA0, false, pB0,
                                                      (short)0, acc[t], false, false);
      acc[t] = __builtin_amdgcn_wmma_f32_16x16x32_f16(false, vA1, false, pB1,
                                                      (short)0, acc[t], false, false);
    }
  }

  // --- epilogue: normalize, gamma*out + x ---
  const float g   = gamma_p[0];
  const float inv = 1.f / lsum;
  #pragma unroll
  for (int t = 0; t < 8; ++t) {
    #pragma unroll
    for (int r = 0; r < 8; ++r) {
      int c = chalf * 128 + t * 16 + r + (hi ? 8 : 0);
      size_t idx = ((size_t)b * NC + c) * NN + i0 + l15;
      out[idx] = g * (acc[t][r] * inv) + x[idx];
    }
  }
}

// ---------------------------------------------------------------------------
extern "C" void kernel_launch(void* const* d_in, const int* in_sizes, int n_in,
                              void* d_out, int out_size, void* d_ws, size_t ws_size,
                              hipStream_t stream) {
  const float* x     = (const float*)d_in[0];
  const float* wf    = (const float*)d_in[1];
  const float* bf    = (const float*)d_in[2];
  const float* wg    = (const float*)d_in[3];
  const float* bg    = (const float*)d_in[4];
  const float* wh    = (const float*)d_in[5];
  const float* bh    = (const float*)d_in[6];
  const float* gamma = (const float*)d_in[7];
  float* out = (float*)d_out;

  _Float16* Qh = (_Float16*)d_ws;                       // 4*4096*32 halves (1 MB)
  _Float16* Kh = Qh + (size_t)NB * NN * ND;             // 1 MB
  _Float16* Vh = Kh + (size_t)NB * NN * ND;             // 8 MB

  proj_kernel<<<dim3(NB * 64), dim3(256), 0, stream>>>(
      x, wf, bf, wg, bg, wh, bh, Qh, Kh, Vh);

  // 2048 wave-tiles, 8 waves per 256-thread block
  attn_kernel<<<dim3(256), dim3(256), 0, stream>>>(
      Qh, Kh, Vh, x, gamma, out);
}
